// LSTM_64673617543229
// MI455X (gfx1250) — matrix operand order repaired and checked by reading, not compiled, not run on hardware
//
#include <hip/hip_runtime.h>
#include <hip/hip_bf16.h>
#include <math.h>

// Problem constants (match reference)
#define T_STEPS 1024
#define BATCH   128
#define DIN     512
#define HDIM    1024
#define OUTD    128

typedef __bf16 bf16_t;
typedef __attribute__((ext_vector_type(16))) __bf16 v16bf;
typedef __attribute__((ext_vector_type(8)))  __bf16 v8bf;
typedef __attribute__((ext_vector_type(8)))  float  v8f;

// ---------------------------------------------------------------------------
// A-fragment loader: 16x32 bf16 A tile, per-lane layout used here (contiguous k = 16*(L/16) + s, same order as B):
//   lane L (row M = m0 + L%16), kb = (L/16)*8:
//     a[0..7]  = A[row][k+2*kb   .. k+2*kb+7]
//     a[8..15] = A[row][k+2*kb+8 .. k+2*kb+15]
// ---------------------------------------------------------------------------
union AFrag { v16bf v; v8bf h[2]; };

__device__ __forceinline__ v16bf load_a_bf16(const bf16_t* __restrict__ arow,
                                             int k, int kb) {
    AFrag u;
    u.h[0] = *(const v8bf*)(arow + k + 2 * kb);
    u.h[1] = *(const v8bf*)(arow + k + 2 * kb + 8);
    return u.v;
}

__device__ __forceinline__ float sigm(float x) {
    return 1.0f / (1.0f + expf(-x));
}

#define WMMA_BF16(A, B, C) \
    __builtin_amdgcn_wmma_f32_16x16x32_bf16(false, (A), false, (B), (short)0, (C), false, false)

// ---------------------------------------------------------------------------
// Fused LSTM cell:  z = A1 @ Wih^T + Hprev @ Whh^T + b_ih + b_hh  (4 gates)
// then c' = sig(f)*c + sig(i)*tanh(g);  h' = sig(o)*tanh(c')  (h' stored bf16)
//
// Block = 1 col tile (16 cols of H) x all 128 batch rows = 4 waves:
//   wave w computes rows [32w, 32w+32) for all 4 gates (8 f32 accumulators).
// Weight (B) tiles are staged through LDS, double buffered:
//   per 32-K chunk the block loads the 4-gate x 16-col x 32-K slice (4 KB)
//   from L2 ONCE (32 B/thread), so per-step weight traffic == weight footprint.
// A-fragments are direct global loads (each row read by exactly one wave).
// Grid: HDIM/16 = 64 blocks x 128 threads.
// ---------------------------------------------------------------------------
template <int K1>
__global__ __launch_bounds__(128)
void lstm_cell_kernel(const bf16_t* __restrict__ A1,     // [BATCH, K1] bf16
                      const bf16_t* __restrict__ Wih,    // [4H, K1] bf16
                      const bf16_t* __restrict__ Hprev,  // [BATCH, H] bf16
                      const bf16_t* __restrict__ Whh,    // [4H, H] bf16
                      const float*  __restrict__ b_ih,   // [4H]
                      const float*  __restrict__ b_hh,   // [4H]
                      float*        __restrict__ c,      // [BATCH, H] f32 in/out
                      bf16_t*       __restrict__ h_out)  // [BATCH, H] bf16 out
{
    // [buf][ (gate*16 + col)*32 + k ]  : 2 x 2048 bf16 = 8 KB
    __shared__ __attribute__((aligned(32))) bf16_t lsB[2][4 * 16 * 32];

    const int tid   = threadIdx.x;
    const int lane  = tid & 31;
    const int rg    = tid >> 5;          // wave id == row group (0..3)
    const int ct    = blockIdx.x;        // col tile (0..63)
    const int m0    = rg * 32;
    const int n0    = ct * 16;
    const int col   = n0 + (lane & 15);  // output column within H
    const int kb    = (lane >> 4) * 8;   // A-frag lane K sub-offset
    const int kboff = (lane >> 4) * 16;  // B-frag lane K sub-offset

    // Cooperative fill mapping: thread moves one 32-byte segment per chunk
    const int fg    = tid >> 5;          // gate 0..3
    const int fcol  = (tid & 31) >> 1;   // col 0..15
    const int fhalf = tid & 1;           // k sub-chunk (0/1)
    const int fofs  = (fg * 16 + fcol) * 32 + fhalf * 16;

    // B-frag LDS offsets for this lane (16 contiguous bf16 per gate)
    const int lofs  = (lane & 15) * 32 + kboff;

    v8f acc00 = {}, acc01 = {}, acc02 = {}, acc03 = {};   // rows m0..m0+15
    v8f acc10 = {}, acc11 = {}, acc12 = {}, acc13 = {};   // rows m0+16..m0+31

    auto run_phase = [&](const bf16_t* __restrict__ Abase,
                         const bf16_t* __restrict__ W, int K) {
        const bf16_t* arow0 = Abase + (size_t)(m0 + (lane & 15)) * K;
        const bf16_t* arow1 = arow0 + (size_t)16 * K;
        const bf16_t* wfill = W + ((size_t)(fg * HDIM + n0 + fcol)) * K + fhalf * 16;
        const int nIter = K / 32;

        // preload chunk 0
        *(v16bf*)(&lsB[0][fofs]) = *(const v16bf*)(wfill);
        __syncthreads();

        for (int it = 0; it < nIter; ++it) {
            const int cur = it & 1;
            const int k   = it * 32;

            // register prefetch of next weight chunk (overlaps WMMAs)
            v16bf pf;
            const bool havePf = (it + 1) < nIter;
            if (havePf) pf = *(const v16bf*)(wfill + (size_t)(it + 1) * 32);

            // B-fragments from LDS (shared across the 4 waves)
            const bf16_t* lb = &lsB[cur][0];
            v16bf b0 = *(const v16bf*)(lb + 0 * 512 + lofs);
            v16bf b1 = *(const v16bf*)(lb + 1 * 512 + lofs);
            v16bf b2 = *(const v16bf*)(lb + 2 * 512 + lofs);
            v16bf b3 = *(const v16bf*)(lb + 3 * 512 + lofs);

            // A-fragments direct from global (rows owned by this wave only)
            v16bf a0 = load_a_bf16(arow0, k, kb);
            v16bf a1 = load_a_bf16(arow1, k, kb);

            acc00 = WMMA_BF16(a0, b0, acc00);
            acc10 = WMMA_BF16(a1, b0, acc10);
            acc01 = WMMA_BF16(a0, b1, acc01);
            acc11 = WMMA_BF16(a1, b1, acc11);
            acc02 = WMMA_BF16(a0, b2, acc02);
            acc12 = WMMA_BF16(a1, b2, acc12);
            acc03 = WMMA_BF16(a0, b3, acc03);
            acc13 = WMMA_BF16(a1, b3, acc13);

            // Store next chunk into the other buffer. Its last readers were
            // fenced by the PREVIOUS barrier, so one barrier per iter suffices.
            if (havePf) *(v16bf*)(&lsB[1 - cur][fofs]) = pf;
            __syncthreads();
        }
    };

    run_phase(A1,    Wih, K1);    // x_t (or h1_t) @ Wih^T
    run_phase(Hprev, Whh, HDIM);  // h_{t-1}       @ Whh^T

    // ---- bias + pointwise LSTM cell (owner-computes: safe in-place on c) ----
    const float bi = b_ih[0 * HDIM + col] + b_hh[0 * HDIM + col];
    const float bf = b_ih[1 * HDIM + col] + b_hh[1 * HDIM + col];
    const float bg = b_ih[2 * HDIM + col] + b_hh[2 * HDIM + col];
    const float bo = b_ih[3 * HDIM + col] + b_hh[3 * HDIM + col];
    const int mhi = (lane >> 4) * 8;   // C/D layout: M = base + v + 8*(lane/16)
#pragma unroll
    for (int v = 0; v < 8; ++v) {
        {   // row tile 0
            const int m = m0 + v + mhi;
            const size_t idx = (size_t)m * HDIM + col;
            const float ig = sigm(acc00[v] + bi);
            const float fg2 = sigm(acc01[v] + bf);
            const float gg = tanhf(acc02[v] + bg);
            const float og = sigm(acc03[v] + bo);
            const float cn = fg2 * c[idx] + ig * gg;
            c[idx]     = cn;
            h_out[idx] = (bf16_t)(og * tanhf(cn));
        }
        {   // row tile 1
            const int m = m0 + 16 + v + mhi;
            const size_t idx = (size_t)m * HDIM + col;
            const float ig = sigm(acc10[v] + bi);
            const float fg2 = sigm(acc11[v] + bf);
            const float gg = tanhf(acc12[v] + bg);
            const float og = sigm(acc13[v] + bo);
            const float cn = fg2 * c[idx] + ig * gg;
            c[idx]     = cn;
            h_out[idx] = (bf16_t)(og * tanhf(cn));
        }
    }
}

// ---------------------------------------------------------------------------
// Output projection: y = h2 @ W_out^T + b_out   [128,128] = [128,1024]x[1024,128]
// Weights are only 0.25 MB -> direct loads are fine. 64 waves, 16 blocks.
// ---------------------------------------------------------------------------
__global__ __launch_bounds__(128)
void proj_kernel(const bf16_t* __restrict__ h2,    // [BATCH, H] bf16
                 const bf16_t* __restrict__ Wout,  // [O, H] bf16
                 const float*  __restrict__ b_out, // [O]
                 float*        __restrict__ y)     // [BATCH, O] f32
{
    const int lane  = threadIdx.x & 31;
    const int wid   = (blockIdx.x * blockDim.x + threadIdx.x) >> 5; // 0..63
    const int rt    = wid >> 3;   // 8 row tiles
    const int ct    = wid & 7;    // 8 col tiles
    const int m0    = rt * 16;
    const int n0    = ct * 16;
    const int col   = n0 + (lane & 15);
    const int row   = m0 + (lane & 15);
    const int kb    = (lane >> 4) * 8;
    const int kboff = (lane >> 4) * 16;

    v8f acc = {};
    const bf16_t* arow = h2 + (size_t)row * HDIM;
    const bf16_t* wr   = Wout + (size_t)col * HDIM + kboff;
#pragma unroll 4
    for (int k = 0; k < HDIM; k += 32) {
        v16bf a = load_a_bf16(arow, k, kb);
        v16bf b = *(const v16bf*)(wr + k);
        acc = WMMA_BF16(a, b, acc);
    }

    const float bias = b_out[col];
    const int mhi = (lane >> 4) * 8;
#pragma unroll
    for (int v = 0; v < 8; ++v) {
        const int m = m0 + v + mhi;
        y[(size_t)m * OUTD + col] = acc[v] + bias;
    }
}

// ---------------------------------------------------------------------------
// One-time helpers: f32 -> bf16 conversion, state zeroing.
// ---------------------------------------------------------------------------
__global__ void cvt_bf16_kernel(const float* __restrict__ src,
                                bf16_t* __restrict__ dst, int n) {
    int i = blockIdx.x * blockDim.x + threadIdx.x;
    if (i < n) dst[i] = (bf16_t)src[i];
}

__global__ void zero_f32_kernel(float* __restrict__ p, int n) {
    int i = blockIdx.x * blockDim.x + threadIdx.x;
    if (i < n) p[i] = 0.0f;
}

__global__ void zero_bf16_kernel(bf16_t* __restrict__ p, int n) {
    int i = blockIdx.x * blockDim.x + threadIdx.x;
    if (i < n) p[i] = (bf16_t)0.0f;
}

// ---------------------------------------------------------------------------
// Host launch
// ---------------------------------------------------------------------------
extern "C" void kernel_launch(void* const* d_in, const int* in_sizes, int n_in,
                              void* d_out, int out_size, void* d_ws, size_t ws_size,
                              hipStream_t stream) {
    const float* x     = (const float*)d_in[0];   // [T, B, D]
    const float* W_ih1 = (const float*)d_in[1];   // [4H, D]
    const float* W_hh1 = (const float*)d_in[2];   // [4H, H]
    const float* b_ih1 = (const float*)d_in[3];
    const float* b_hh1 = (const float*)d_in[4];
    const float* W_ih2 = (const float*)d_in[5];   // [4H, H]
    const float* W_hh2 = (const float*)d_in[6];   // [4H, H]
    const float* b_ih2 = (const float*)d_in[7];
    const float* b_hh2 = (const float*)d_in[8];
    const float* W_out = (const float*)d_in[9];   // [O, H]
    const float* b_out = (const float*)d_in[10];
    float* y = (float*)d_out;                     // [T, B, O]

    // Workspace layout
    char* ws = (char*)d_ws;
    const size_t n_wih1 = (size_t)4 * HDIM * DIN;
    const size_t n_whh1 = (size_t)4 * HDIM * HDIM;
    const size_t n_wih2 = (size_t)4 * HDIM * HDIM;
    const size_t n_whh2 = (size_t)4 * HDIM * HDIM;
    const size_t n_wout = (size_t)OUTD * HDIM;
    const size_t n_x    = (size_t)T_STEPS * BATCH * DIN;
    const size_t n_state = (size_t)BATCH * HDIM;

    size_t off = 0;
    bf16_t* wih1b = (bf16_t*)(ws + off); off += n_wih1 * sizeof(bf16_t);
    bf16_t* whh1b = (bf16_t*)(ws + off); off += n_whh1 * sizeof(bf16_t);
    bf16_t* wih2b = (bf16_t*)(ws + off); off += n_wih2 * sizeof(bf16_t);
    bf16_t* whh2b = (bf16_t*)(ws + off); off += n_whh2 * sizeof(bf16_t);
    bf16_t* woutb = (bf16_t*)(ws + off); off += n_wout * sizeof(bf16_t);
    bf16_t* xb    = (bf16_t*)(ws + off); off += n_x    * sizeof(bf16_t);
    bf16_t* hstates = (bf16_t*)(ws + off); off += 4 * n_state * sizeof(bf16_t);
    bf16_t* h1a = hstates + 0 * n_state;
    bf16_t* h1b = hstates + 1 * n_state;
    bf16_t* h2a = hstates + 2 * n_state;
    bf16_t* h2b = hstates + 3 * n_state;
    float* cstates = (float*)(ws + off); off += 2 * n_state * sizeof(float);
    float* c1 = cstates + 0 * n_state;
    float* c2 = cstates + 1 * n_state;

    // One-time per launch: convert weights + whole input tensor to bf16
    const int CT = 256;
    cvt_bf16_kernel<<<(int)((n_wih1 + CT - 1) / CT), CT, 0, stream>>>(W_ih1, wih1b, (int)n_wih1);
    cvt_bf16_kernel<<<(int)((n_whh1 + CT - 1) / CT), CT, 0, stream>>>(W_hh1, whh1b, (int)n_whh1);
    cvt_bf16_kernel<<<(int)((n_wih2 + CT - 1) / CT), CT, 0, stream>>>(W_ih2, wih2b, (int)n_wih2);
    cvt_bf16_kernel<<<(int)((n_whh2 + CT - 1) / CT), CT, 0, stream>>>(W_hh2, whh2b, (int)n_whh2);
    cvt_bf16_kernel<<<(int)((n_wout + CT - 1) / CT), CT, 0, stream>>>(W_out, woutb, (int)n_wout);
    cvt_bf16_kernel<<<(int)((n_x    + CT - 1) / CT), CT, 0, stream>>>(x, xb, (int)n_x);

    // Zero all states every call (determinism; harness does not re-poison)
    zero_bf16_kernel<<<(int)((4 * n_state + CT - 1) / CT), CT, 0, stream>>>(hstates, (int)(4 * n_state));
    zero_f32_kernel<<<(int)((2 * n_state + CT - 1) / CT), CT, 0, stream>>>(cstates, (int)(2 * n_state));

    // Sequential scan over T: 3 kernels per timestep
    const int CELL_BLOCKS = HDIM / 16;                       // 64 blocks (1 col tile each)
    const int PROJ_BLOCKS = (BATCH / 16) * (OUTD / 16) / 4;  // 16 blocks x 4 waves
    for (int t = 0; t < T_STEPS; ++t) {
        const bf16_t* h1p = (t & 1) ? h1b : h1a;
        bf16_t*       h1c = (t & 1) ? h1a : h1b;
        const bf16_t* h2p = (t & 1) ? h2b : h2a;
        bf16_t*       h2c = (t & 1) ? h2a : h2b;

        const bf16_t* x_t = xb + (size_t)t * BATCH * DIN;

        lstm_cell_kernel<DIN><<<CELL_BLOCKS, 128, 0, stream>>>(
            x_t, wih1b, h1p, whh1b, b_ih1, b_hh1, c1, h1c);

        lstm_cell_kernel<HDIM><<<CELL_BLOCKS, 128, 0, stream>>>(
            h1c, wih2b, h2p, whh2b, b_ih2, b_hh2, c2, h2c);

        proj_kernel<<<PROJ_BLOCKS, 128, 0, stream>>>(
            h2c, woutb, b_out, y + (size_t)t * BATCH * OUTD);
    }
}
